// MambaAttnBlock_20332375180000
// MI455X (gfx1250) — compile-verified, matrix operand order checked by reading, not run on hardware
//
#include <hip/hip_runtime.h>
#include <hip/hip_bf16.h>
#include <cstdint>
#include <cstddef>

// Problem constants
#define BB 4
#define LL 3136
#define DD 256
#define DHH 256
#define MTOK (BB * LL)   // 12544 tokens; 784 m-tiles of 16; 2 per wave; 8 waves -> 49 blocks

typedef __attribute__((ext_vector_type(16))) _Float16 v16h;
typedef __attribute__((ext_vector_type(8)))  _Float16 v8h;
typedef __attribute__((ext_vector_type(8)))  float    v8f;

// ---------------------------------------------------------------------------
// WMMA helpers (CDNA5 16x16x32 f16 -> f32)
// A operand layout (16x32, 16-bit): lane(0..15)=row M, lane(16..31)=row M with
// K offset +8; per lane 16 halfs = K in {g*8 .. g*8+7} and {16+g*8 .. 16+g*8+7}.
// B operand (32x16) uses the same per-lane pattern with lane = column N, which
// for C = A * W^T (W row-major (N,K)) means lane reads W row n along K.
// ---------------------------------------------------------------------------
__device__ __forceinline__ v16h load_frag16(const _Float16* __restrict__ base, int ld) {
  const int lane = threadIdx.x & 31;
  const int row  = lane & 15;
  const int g    = lane >> 4;           // 0 or 1
  const _Float16* p = base + (size_t)row * ld + g * 8;
  v8h lo = *(const v8h*)(p);            // K = g*8 .. g*8+7
  v8h hi = *(const v8h*)(p + 16);       // K = 16+g*8 .. 16+g*8+7
  v16h r;
#pragma unroll
  for (int i = 0; i < 8; ++i) { r[i] = lo[i]; r[8 + i] = hi[i]; }
  return r;
}

// Register-blocked GEMM core: MT m-tiles x NT n-tiles per wave.
template <int MT, int NT>
__device__ __forceinline__ void gemm_acc(const _Float16* __restrict__ A, int lda,
                                         const _Float16* __restrict__ W, int ldw,
                                         int K, int m0, int n0, v8f acc[MT][NT]) {
  for (int k0 = 0; k0 < K; k0 += 32) {
    if (k0 + 32 < K) {
      // hint next B rows toward the caches (global_prefetch_b8)
      __builtin_prefetch((const void*)(W + (size_t)(n0 + ((threadIdx.x & 31) & (NT * 16 - 1))) * ldw + k0 + 32), 0, 1);
    }
    v16h a[MT];
#pragma unroll
    for (int i = 0; i < MT; ++i)
      a[i] = load_frag16(A + (size_t)(m0 + i * 16) * lda + k0, lda);
#pragma unroll
    for (int t = 0; t < NT; ++t) {
      v16h b = load_frag16(W + (size_t)(n0 + t * 16) * ldw + k0, ldw);
#pragma unroll
      for (int i = 0; i < MT; ++i)
        acc[i][t] = __builtin_amdgcn_wmma_f32_16x16x32_f16(false, a[i], false, b,
                                                           (short)0, acc[i][t], false, false);
    }
  }
}

// C/D tile layout: lane n = lane&15 ; row m = (lane>>4)*8 + r ; r = VGPR index.

// ---------------------------------------------------------------------------
// Misc small kernels
// ---------------------------------------------------------------------------
__global__ void k_cvt_f16(const float* __restrict__ in, _Float16* __restrict__ out, int n) {
  int i = blockIdx.x * 256 + threadIdx.x;
  if (i < n) out[i] = (_Float16)in[i];
}

// One wave32 per row of 256; 8 rows per block.
__global__ void k_layernorm(const float* __restrict__ X, const float* __restrict__ g,
                            const float* __restrict__ b, _Float16* __restrict__ Y) {
  const int row  = blockIdx.x * 8 + (threadIdx.x >> 5);
  const int lane = threadIdx.x & 31;
  const float* xr = X + (size_t)row * DD;
  float v[8];
  float s = 0.f;
#pragma unroll
  for (int i = 0; i < 8; ++i) { v[i] = xr[lane + i * 32]; s += v[i]; }
#pragma unroll
  for (int off = 16; off; off >>= 1) s += __shfl_xor(s, off, 32);
  const float mean = s * (1.0f / 256.0f);
  float s2 = 0.f;
#pragma unroll
  for (int i = 0; i < 8; ++i) { float d = v[i] - mean; s2 += d * d; }
#pragma unroll
  for (int off = 16; off; off >>= 1) s2 += __shfl_xor(s2, off, 32);
  const float rstd = rsqrtf(s2 * (1.0f / 256.0f) + 1e-5f);
#pragma unroll
  for (int i = 0; i < 8; ++i) {
    const int c = lane + i * 32;
    Y[(size_t)row * DD + c] = (_Float16)((v[i] - mean) * rstd * g[c] + b[c]);
  }
}

// Depthwise conv3 (SAME) + SiLU over xz (MTOK x 512, token-major).
// e<256 -> xb path (f16 for x_proj GEMM + f32 u for scan); e>=256 -> zb into ycat[:,256:].
__global__ void k_conv_silu(const float* __restrict__ XZ,
                            const float* __restrict__ wx, const float* __restrict__ wz,
                            _Float16* __restrict__ xc, float* __restrict__ u,
                            _Float16* __restrict__ ycat) {
  const size_t idx = (size_t)blockIdx.x * 256 + threadIdx.x;
  if (idx >= (size_t)MTOK * 512) return;
  const int    e = (int)(idx & 511);
  const size_t m = idx >> 9;
  const int    l = (int)(m % LL);
  float w0, w1, w2;
  if (e < 256) { w0 = wx[e * 3]; w1 = wx[e * 3 + 1]; w2 = wx[e * 3 + 2]; }
  else { const int ez = e - 256; w0 = wz[ez * 3]; w1 = wz[ez * 3 + 1]; w2 = wz[ez * 3 + 2]; }
  const float xm = (l > 0)      ? XZ[(m - 1) * 512 + e] : 0.f;
  const float x0 =                XZ[m * 512 + e];
  const float xp = (l < LL - 1) ? XZ[(m + 1) * 512 + e] : 0.f;
  const float vv = w0 * xm + w1 * x0 + w2 * xp;
  const float s  = vv / (1.f + __expf(-vv));
  if (e < 256) { xc[m * 256 + e] = (_Float16)s; u[m * 256 + e] = s; }
  else         { ycat[m * 512 + e] = (_Float16)s; }
}

// Selective scan: one 16-lane half-wave per (b,d) channel; lane = state index n.
// delta projection (K=16) fused via shuffle reduction; y reduction likewise.
__global__ void k_scan(const float* __restrict__ x_dbl,   // (MTOK,48)
                       const float* __restrict__ u,       // (MTOK,256)
                       const float* __restrict__ A_log,   // (256,16)
                       const float* __restrict__ dtw,     // (256,16)
                       const float* __restrict__ dtb,     // (256)
                       const float* __restrict__ Dp,      // (256)
                       _Float16* __restrict__ ycat) {     // (MTOK,512) cols 0..255
  const int p = (blockIdx.x * 256 + threadIdx.x) >> 4;    // (b,d) pair, 0..1023
  const int n = threadIdx.x & 15;
  const int b = p / DHH;
  const int d = p % DHH;
  const float A_dn  = -__expf(A_log[d * 16 + n]);
  const float dtw_n = dtw[d * 16 + n];
  const float bias  = dtb[d];
  const float Dd    = Dp[d];
  float h = 0.f;
  for (int l = 0; l < LL; ++l) {
    const size_t m = (size_t)b * LL + l;
    const float* xd = x_dbl + m * 48;
    float dsum = xd[n] * dtw_n;
    dsum += __shfl_xor(dsum, 1, 32);
    dsum += __shfl_xor(dsum, 2, 32);
    dsum += __shfl_xor(dsum, 4, 32);
    dsum += __shfl_xor(dsum, 8, 32);
    const float draw  = dsum + bias;
    const float delta = (draw > 20.f) ? draw : log1pf(__expf(draw));
    const float uval  = u[m * 256 + d];
    const float Bn = xd[16 + n];
    const float Cn = xd[32 + n];
    h = __expf(delta * A_dn) * h + delta * Bn * uval;
    float y = h * Cn;
    y += __shfl_xor(y, 1, 32);
    y += __shfl_xor(y, 2, 32);
    y += __shfl_xor(y, 4, 32);
    y += __shfl_xor(y, 8, 32);
    if (n == 0) ycat[m * 512 + d] = (_Float16)(y + uval * Dd);
  }
}

// ---------------------------------------------------------------------------
// WMMA GEMM kernels (block = 8 waves; wave => 32x64 output; grid.x = MTOK/256)
// ---------------------------------------------------------------------------
__global__ void k_gemm_inproj(const _Float16* __restrict__ A, const _Float16* __restrict__ W,
                              float* __restrict__ C) { // (MTOK,512) f32
  const int mt0 = (blockIdx.x * 8 + (threadIdx.x >> 5)) * 2;
  const int n0  = blockIdx.y * 64;
  v8f acc[2][4] = {};
  gemm_acc<2, 4>(A, 256, W, 256, 256, mt0 * 16, n0, acc);
  const int lane = threadIdx.x & 31;
  const int n = lane & 15, gm = (lane >> 4) * 8;
#pragma unroll
  for (int i = 0; i < 2; ++i)
#pragma unroll
    for (int t = 0; t < 4; ++t)
#pragma unroll
      for (int r = 0; r < 8; ++r)
        __builtin_nontemporal_store(
            acc[i][t][r],
            &C[(size_t)((mt0 + i) * 16 + gm + r) * 512 + n0 + t * 16 + n]);
}

__global__ void k_gemm_xproj(const _Float16* __restrict__ A, const _Float16* __restrict__ W,
                             float* __restrict__ C) { // (MTOK,48) f32
  const int mt0 = (blockIdx.x * 8 + (threadIdx.x >> 5)) * 2;
  v8f acc[2][3] = {};
  gemm_acc<2, 3>(A, 256, W, 256, 256, mt0 * 16, 0, acc);
  const int lane = threadIdx.x & 31;
  const int n = lane & 15, gm = (lane >> 4) * 8;
#pragma unroll
  for (int i = 0; i < 2; ++i)
#pragma unroll
    for (int t = 0; t < 3; ++t)
#pragma unroll
      for (int r = 0; r < 8; ++r)
        C[(size_t)((mt0 + i) * 16 + gm + r) * 48 + t * 16 + n] = acc[i][t][r];
}

__global__ void k_gemm_outproj(const _Float16* __restrict__ A, const _Float16* __restrict__ W,
                               const float* __restrict__ resid, float* __restrict__ C) { // (MTOK,256)
  const int mt0 = (blockIdx.x * 8 + (threadIdx.x >> 5)) * 2;
  const int n0  = blockIdx.y * 64;
  v8f acc[2][4] = {};
  gemm_acc<2, 4>(A, 512, W, 512, 512, mt0 * 16, n0, acc);
  const int lane = threadIdx.x & 31;
  const int n = lane & 15, gm = (lane >> 4) * 8;
#pragma unroll
  for (int i = 0; i < 2; ++i)
#pragma unroll
    for (int t = 0; t < 4; ++t)
#pragma unroll
      for (int r = 0; r < 8; ++r) {
        const size_t o = (size_t)((mt0 + i) * 16 + gm + r) * 256 + n0 + t * 16 + n;
        C[o] = resid[o] + acc[i][t][r];
      }
}

__global__ void k_gemm_mlp1(const _Float16* __restrict__ A, const _Float16* __restrict__ W,
                            const float* __restrict__ b1, _Float16* __restrict__ H) { // (MTOK,1024) f16
  const int mt0 = (blockIdx.x * 8 + (threadIdx.x >> 5)) * 2;
  const int n0  = blockIdx.y * 64;
  v8f acc[2][4] = {};
  gemm_acc<2, 4>(A, 256, W, 256, 256, mt0 * 16, n0, acc);
  const int lane = threadIdx.x & 31;
  const int n = lane & 15, gm = (lane >> 4) * 8;
#pragma unroll
  for (int i = 0; i < 2; ++i)
#pragma unroll
    for (int t = 0; t < 4; ++t) {
      const float bias = b1[n0 + t * 16 + n];
#pragma unroll
      for (int r = 0; r < 8; ++r) {
        const float v = acc[i][t][r] + bias;
        const float gelu = 0.5f * v * (1.0f + erff(v * 0.70710678118654752f));
        H[(size_t)((mt0 + i) * 16 + gm + r) * 1024 + n0 + t * 16 + n] = (_Float16)gelu;
      }
    }
}

__global__ void k_gemm_mlp2(const _Float16* __restrict__ A, const _Float16* __restrict__ W,
                            const float* __restrict__ b2, const float* __restrict__ resid,
                            float* __restrict__ Out) { // (MTOK,256) f32 final
  const int mt0 = (blockIdx.x * 8 + (threadIdx.x >> 5)) * 2;
  const int n0  = blockIdx.y * 64;
  v8f acc[2][4] = {};
  gemm_acc<2, 4>(A, 1024, W, 1024, 1024, mt0 * 16, n0, acc);
  const int lane = threadIdx.x & 31;
  const int n = lane & 15, gm = (lane >> 4) * 8;
#pragma unroll
  for (int i = 0; i < 2; ++i)
#pragma unroll
    for (int t = 0; t < 4; ++t) {
      const float bias = b2[n0 + t * 16 + n];
#pragma unroll
      for (int r = 0; r < 8; ++r) {
        const size_t o = (size_t)((mt0 + i) * 16 + gm + r) * 256 + n0 + t * 16 + n;
        Out[o] = resid[o] + bias + acc[i][t][r];
      }
    }
}

// ---------------------------------------------------------------------------
// Host launch
// ---------------------------------------------------------------------------
extern "C" void kernel_launch(void* const* d_in, const int* in_sizes, int n_in,
                              void* d_out, int out_size, void* d_ws, size_t ws_size,
                              hipStream_t stream) {
  const float* x        = (const float*)d_in[0];
  const float* ln1_g    = (const float*)d_in[1];
  const float* ln1_b    = (const float*)d_in[2];
  const float* ln2_g    = (const float*)d_in[3];
  const float* ln2_b    = (const float*)d_in[4];
  const float* in_proj  = (const float*)d_in[5];   // (512,256)
  const float* conv_x_w = (const float*)d_in[6];   // (256,1,3)
  const float* conv_z_w = (const float*)d_in[7];
  const float* x_proj   = (const float*)d_in[8];   // (48,256)
  const float* dt_projw = (const float*)d_in[9];   // (256,16)
  const float* dt_projb = (const float*)d_in[10];  // (256)
  const float* A_log    = (const float*)d_in[11];  // (256,16)
  const float* Dp       = (const float*)d_in[12];  // (256)
  const float* out_proj = (const float*)d_in[13];  // (256,512)
  const float* mlp_w1   = (const float*)d_in[14];  // (1024,256)
  const float* mlp_b1   = (const float*)d_in[15];
  const float* mlp_w2   = (const float*)d_in[16];  // (256,1024)
  const float* mlp_b2   = (const float*)d_in[17];
  float* out = (float*)d_out;

  char* ws = (char*)d_ws;
  size_t off = 0;
  auto alloc = [&](size_t bytes) -> char* {
    off = (off + 255) & ~(size_t)255;
    char* p = ws + off;
    off += bytes;
    return p;
  };

  _Float16* w_inproj_h  = (_Float16*)alloc((size_t)512 * 256 * 2);
  _Float16* w_xproj_h   = (_Float16*)alloc((size_t)48 * 256 * 2);
  _Float16* w_outproj_h = (_Float16*)alloc((size_t)256 * 512 * 2);
  _Float16* w_mlp1_h    = (_Float16*)alloc((size_t)1024 * 256 * 2);
  _Float16* w_mlp2_h    = (_Float16*)alloc((size_t)256 * 1024 * 2);
  _Float16* xn_h        = (_Float16*)alloc((size_t)MTOK * 256 * 2);  // LN1 out; reused for LN2 out
  float*    xz_f        = (float*)   alloc((size_t)MTOK * 512 * 4);  // in_proj out; region reused for mlp hidden f16
  _Float16* xc_h        = (_Float16*)alloc((size_t)MTOK * 256 * 2);  // silu(conv(xb)) f16
  float*    u_f         = (float*)   alloc((size_t)MTOK * 256 * 4);  // silu(conv(xb)) f32; reused as out1 f32
  _Float16* ycat_h      = (_Float16*)alloc((size_t)MTOK * 512 * 2);  // [y | zb] f16
  float*    xdbl_f      = (float*)   alloc((size_t)MTOK * 48 * 4);   // x_proj out
  _Float16* hh_h        = (_Float16*)xz_f;                           // mlp hidden (MTOK,1024) f16 aliases xz
  float*    out1_f      = u_f;                                       // post-mamba residual aliases u

  // 1) weights -> f16
  k_cvt_f16<<<(512 * 256 + 255) / 256, 256, 0, stream>>>(in_proj, w_inproj_h, 512 * 256);
  k_cvt_f16<<<(48 * 256 + 255) / 256, 256, 0, stream>>>(x_proj, w_xproj_h, 48 * 256);
  k_cvt_f16<<<(256 * 512 + 255) / 256, 256, 0, stream>>>(out_proj, w_outproj_h, 256 * 512);
  k_cvt_f16<<<(1024 * 256 + 255) / 256, 256, 0, stream>>>(mlp_w1, w_mlp1_h, 1024 * 256);
  k_cvt_f16<<<(256 * 1024 + 255) / 256, 256, 0, stream>>>(mlp_w2, w_mlp2_h, 256 * 1024);

  // 2) LN1
  k_layernorm<<<MTOK / 8, 256, 0, stream>>>(x, ln1_g, ln1_b, xn_h);

  // 3) in_proj GEMM: (MTOK,512) = xn @ W^T
  k_gemm_inproj<<<dim3(MTOK / 256, 8), 256, 0, stream>>>(xn_h, w_inproj_h, xz_f);

  // 4) depthwise conv3 + SiLU
  k_conv_silu<<<(int)(((size_t)MTOK * 512 + 255) / 256), 256, 0, stream>>>(
      xz_f, conv_x_w, conv_z_w, xc_h, u_f, ycat_h);

  // NOTE: xz_f region is dead after conv; safe to alias for mlp hidden later.

  // 5) x_proj GEMM: (MTOK,48)
  k_gemm_xproj<<<dim3(MTOK / 256, 1), 256, 0, stream>>>(xc_h, w_xproj_h, xdbl_f);

  // 6) selective scan -> ycat[:, 0:256]
  k_scan<<<(BB * DHH * 16) / 256, 256, 0, stream>>>(xdbl_f, u_f, A_log, dt_projw,
                                                    dt_projb, Dp, ycat_h);

  // 7) out_proj GEMM + residual -> out1 (aliases u_f; u dead after scan)
  k_gemm_outproj<<<dim3(MTOK / 256, 4), 256, 0, stream>>>(ycat_h, w_outproj_h, x, out1_f);

  // 8) LN2 -> xn_h (reuse)
  k_layernorm<<<MTOK / 8, 256, 0, stream>>>(out1_f, ln2_g, ln2_b, xn_h);

  // 9) mlp1 GEMM + bias + GELU -> hh (f16, aliases xz region)
  k_gemm_mlp1<<<dim3(MTOK / 256, 16), 256, 0, stream>>>(xn_h, w_mlp1_h, mlp_b1, hh_h);

  // 10) mlp2 GEMM + bias + residual -> d_out
  k_gemm_mlp2<<<dim3(MTOK / 256, 4), 256, 0, stream>>>(hh_h, w_mlp2_h, mlp_b2, out1_f, out);
}
